// MultiheadAttention_40776419508884
// MI455X (gfx1250) — matrix operand order repaired and checked by reading, not compiled, not run on hardware
//
#include <hip/hip_runtime.h>
#include <hip/hip_bf16.h>

typedef __bf16 bf16;
typedef __attribute__((ext_vector_type(16))) __bf16 v16bf;
typedef __attribute__((ext_vector_type(8)))  float   v8f;

union Frag {
    v16bf v;
    uint4 u[2];
};

__device__ __forceinline__ v8f wmma_bf16(const Frag& a, const Frag& b, v8f c) {
    // D = A(16x32 bf16) * B(32x16 bf16) + C(16x16 f32)
    return __builtin_amdgcn_wmma_f32_16x16x32_bf16(
        /*neg_a=*/false, a.v, /*neg_b=*/false, b.v,
        /*c_mod=*/(short)0, c, /*reuse_a=*/false, /*reuse_b=*/false);
}

// CDNA5 async global->LDS copy (16B per lane), tracked by ASYNCcnt.
// ldsaddr: per-lane LDS byte address (low 32 bits of generic LDS pointer).
// gaddr:   per-lane 64-bit global address (VGPR pair).
__device__ __forceinline__ void async_b128(unsigned ldsaddr, const void* gaddr) {
    asm volatile("global_load_async_to_lds_b128 %0, %1, off"
                 :: "v"(ldsaddr), "v"(gaddr)
                 : "memory");
}
__device__ __forceinline__ void wait_async0() {
    asm volatile("s_wait_asynccnt 0" ::: "memory");
}

// ---------------------------------------------------------------------------
// fp32 -> bf16 conversion (4 elements / thread)
// ---------------------------------------------------------------------------
__global__ __launch_bounds__(256) void cvt_f32_bf16(const float* __restrict__ in,
                                                    bf16* __restrict__ out,
                                                    int n) {
    int i = (blockIdx.x * 256 + threadIdx.x) * 4;
    if (i + 3 < n) {
        float4 f = *(const float4*)(in + i);
        union { bf16 h[4]; uint2 u; } pk;
        pk.h[0] = (bf16)f.x; pk.h[1] = (bf16)f.y;
        pk.h[2] = (bf16)f.z; pk.h[3] = (bf16)f.w;
        *(uint2*)(out + i) = pk.u;
    }
}

// ---------------------------------------------------------------------------
// GEMM: out[m,n] = sum_k A[m,k] * W[n,k] + bias[n]     (M=4096, N=K=1024)
//   mode 0: write bf16, head-split layout [B,H,S,64], scaled by outScale
//   mode 1: write fp32, row-major [M,1024]
// Macro tile 128x128x32; 8 waves; each wave 4(M) x 2(N) WMMA tiles.
// Double-buffered LDS staging via async global->LDS copies (ASYNCcnt).
// ---------------------------------------------------------------------------
__global__ __launch_bounds__(256) void proj_kernel(const bf16* __restrict__ A,
                                                   const bf16* __restrict__ W,
                                                   const float* __restrict__ bias,
                                                   bf16* __restrict__ outHead,
                                                   float* __restrict__ outF,
                                                   float outScale, int mode) {
    __shared__ __align__(16) bf16 As[2][128 * 40];
    __shared__ __align__(16) bf16 Bs[2][128 * 40];

    const int tid  = threadIdx.x;
    const int lane = tid & 31;
    const int wave = tid >> 5;
    const int hf   = lane >> 4;    // lane half (0/1)
    const int l16  = lane & 15;
    const int wm   = wave >> 2;    // 0..1  -> 64 rows each
    const int wn   = wave & 3;     // 0..3  -> 32 cols each
    const int mBase = blockIdx.y * 128;
    const int nBase = blockIdx.x * 128;

    // per-thread staging coordinates: two 16B chunks per matrix per tile
    const int sr0 = tid >> 2;            // rows 0..63   (pass 0)
    const int sr1 = (256 + tid) >> 2;    // rows 64..127 (pass 1)
    const int sc  = (tid & 3) * 8;

    auto stage = [&](int buf, int k0) {
        async_b128((unsigned)(uintptr_t)&As[buf][sr0 * 40 + sc],
                   &A[(size_t)(mBase + sr0) * 1024 + k0 + sc]);
        async_b128((unsigned)(uintptr_t)&As[buf][sr1 * 40 + sc],
                   &A[(size_t)(mBase + sr1) * 1024 + k0 + sc]);
        async_b128((unsigned)(uintptr_t)&Bs[buf][sr0 * 40 + sc],
                   &W[(size_t)(nBase + sr0) * 1024 + k0 + sc]);
        async_b128((unsigned)(uintptr_t)&Bs[buf][sr1 * 40 + sc],
                   &W[(size_t)(nBase + sr1) * 1024 + k0 + sc]);
    };

    v8f acc[4][2];
    #pragma unroll
    for (int mi = 0; mi < 4; mi++)
        #pragma unroll
        for (int ni = 0; ni < 2; ni++)
            acc[mi][ni] = {};

    stage(0, 0);

    for (int k0 = 0; k0 < 1024; k0 += 32) {
        const int buf = (k0 >> 5) & 1;
        wait_async0();        // this wave's portion of tile `buf` landed in LDS
        __syncthreads();      // everyone's portion landed; prev compute done
        if (k0 + 32 < 1024) stage(buf ^ 1, k0 + 32);  // prefetch next tile

        Frag af[4], bf_[2];
        #pragma unroll
        for (int mi = 0; mi < 4; mi++) {
            // A layout (16x32 bf16): lane<16: K {0..7,16..23}; lane>=16: {8..15,24..31}
            const bf16* p0 = &As[buf][(wm * 64 + mi * 16 + l16) * 40 + hf * 8];
            af[mi].u[0] = *(const uint4*)p0;
            af[mi].u[1] = *(const uint4*)(p0 + 16);
        }
        #pragma unroll
        for (int ni = 0; ni < 2; ni++) {
            // B layout (32x16 bf16): lane holds col n=l16; lane<16: K {0..7,16..23}; lane>=16: {8..15,24..31}
            const bf16* p0 = &Bs[buf][(wn * 32 + ni * 16 + l16) * 40 + hf * 8];
            bf_[ni].u[0] = *(const uint4*)p0;
            bf_[ni].u[1] = *(const uint4*)(p0 + 16);
        }
        #pragma unroll
        for (int mi = 0; mi < 4; mi++)
            #pragma unroll
            for (int ni = 0; ni < 2; ni++)
                acc[mi][ni] = wmma_bf16(af[mi], bf_[ni], acc[mi][ni]);
    }

    // Epilogue. C layout: VGPR r -> row M = r + 8*half, col N = l16.
    #pragma unroll
    for (int mi = 0; mi < 4; mi++) {
        #pragma unroll
        for (int ni = 0; ni < 2; ni++) {
            int n = nBase + wn * 32 + ni * 16 + l16;
            float bv = bias[n];
            #pragma unroll
            for (int r = 0; r < 8; r++) {
                int m = mBase + wm * 64 + mi * 16 + r + hf * 8;
                float val = (acc[mi][ni][r] + bv) * outScale;
                if (mode == 0) {
                    int b = m >> 11, s = m & 2047;
                    int h = n >> 6,  d = n & 63;
                    outHead[(((size_t)b * 16 + h) * 2048 + s) * 64 + d] = (bf16)val;
                } else {
                    outF[(size_t)m * 1024 + n] = val;
                }
            }
        }
    }
}

// ---------------------------------------------------------------------------
// Flash attention over heads. Q pre-scaled by 1/sqrt(d_k).
// Grid: 32 (b*h) x 16 (q blocks of 128). 8 waves; each wave = 16 query rows.
// Per iteration: K block (32x64) staged via async global->LDS; V block staged
// transposed (64x32) through VGPRs (transpose needs element scatter).
// ---------------------------------------------------------------------------
__global__ __launch_bounds__(256) void attn_kernel(const bf16* __restrict__ Qh,
                                                   const bf16* __restrict__ Kh,
                                                   const bf16* __restrict__ Vh,
                                                   bf16* __restrict__ ctx) {
    __shared__ __align__(16) bf16 Ks[32 * 72];       // [t][d]
    __shared__ __align__(16) bf16 Vt[64 * 40];       // [d][t]
    __shared__ __align__(16) bf16 Ps[8 * 16 * 40];   // per-wave P scratch [row][t]

    const int tid  = threadIdx.x;
    const int lane = tid & 31;
    const int wave = tid >> 5;
    const int hf   = lane >> 4;
    const int l16  = lane & 15;
    const int bh   = blockIdx.x >> 4;   // 0..31
    const int qblk = blockIdx.x & 15;
    const int qBase = qblk * 128 + wave * 16;

    const bf16* Qb = Qh + (size_t)bh * 2048 * 64;
    const bf16* Kb = Kh + (size_t)bh * 2048 * 64;
    const bf16* Vb = Vh + (size_t)bh * 2048 * 64;

    // Q fragments for d-chunks [0,32) and [32,64)
    Frag qf[2];
    {
        const bf16* qrow = Qb + (size_t)(qBase + l16) * 64 + hf * 8;
        #pragma unroll
        for (int c = 0; c < 2; c++) {
            qf[c].u[0] = *(const uint4*)(qrow + c * 32);
            qf[c].u[1] = *(const uint4*)(qrow + c * 32 + 16);
        }
    }

    v8f cacc[4];
    #pragma unroll
    for (int dq = 0; dq < 4; dq++) cacc[dq] = {};
    float mstate[8], lstate[8];
    #pragma unroll
    for (int r = 0; r < 8; r++) { mstate[r] = -__builtin_inff(); lstate[r] = 0.0f; }

    bf16* PsW = &Ps[wave * 16 * 40];
    const int st = tid >> 3;            // staging row t (0..31)
    const int sd = (tid & 7) * 8;       // staging col d (0..56 step 8)

    for (int tb = 0; tb < 2048; tb += 32) {
        __syncthreads();
        // K block: async global->LDS, 16B per thread
        async_b128((unsigned)(uintptr_t)&Ks[st * 72 + sd],
                   &Kb[(size_t)(tb + st) * 64 + sd]);
        // V block: load 8 bf16 and scatter-transpose into Vt[d][t]
        {
            union { uint4 u; bf16 h[8]; } vv;
            vv.u = *(const uint4*)&Vb[(size_t)(tb + st) * 64 + sd];
            #pragma unroll
            for (int j = 0; j < 8; j++) Vt[(sd + j) * 40 + st] = vv.h[j];
        }
        wait_async0();
        __syncthreads();

        // scores: S0 = Q*K^T (t 0..15), S1 (t 16..31), K-dim = d (two chunks)
        v8f s0 = {}, s1 = {};
        #pragma unroll
        for (int c = 0; c < 2; c++) {
            Frag k0f, k1f;
            const bf16* kp0 = &Ks[l16 * 72 + c * 32 + hf * 8];
            k0f.u[0] = *(const uint4*)kp0;
            k0f.u[1] = *(const uint4*)(kp0 + 16);
            const bf16* kp1 = &Ks[(16 + l16) * 72 + c * 32 + hf * 8];
            k1f.u[0] = *(const uint4*)kp1;
            k1f.u[1] = *(const uint4*)(kp1 + 16);
            s0 = wmma_bf16(qf[c], k0f, s0);
            s1 = wmma_bf16(qf[c], k1f, s1);
        }

        // online softmax, per row slot r (row = r + 8*half for this lane)
        #pragma unroll
        for (int r = 0; r < 8; r++) {
            float v = fmaxf(s0[r], s1[r]);
            #pragma unroll
            for (int mk = 1; mk < 16; mk <<= 1) v = fmaxf(v, __shfl_xor(v, mk, 32));
            float mnew = fmaxf(mstate[r], v);
            float p0 = __expf(s0[r] - mnew);
            float p1 = __expf(s1[r] - mnew);
            float rs = p0 + p1;
            #pragma unroll
            for (int mk = 1; mk < 16; mk <<= 1) rs += __shfl_xor(rs, mk, 32);
            float corr = __expf(mstate[r] - mnew);
            lstate[r] = lstate[r] * corr + rs;
            mstate[r] = mnew;
            #pragma unroll
            for (int dq = 0; dq < 4; dq++) cacc[dq][r] *= corr;
            int row = r + hf * 8;
            PsW[row * 40 + l16]      = (bf16)p0;
            PsW[row * 40 + 16 + l16] = (bf16)p1;
        }

        // ctx += P (16x32) * V (32x64): A-frag from Ps, B-frags from Vt
        Frag pf;
        const bf16* pp = &PsW[l16 * 40 + hf * 8];
        pf.u[0] = *(const uint4*)pp;
        pf.u[1] = *(const uint4*)(pp + 16);
        #pragma unroll
        for (int dq = 0; dq < 4; dq++) {
            Frag vf;
            const bf16* vp = &Vt[(dq * 16 + l16) * 40 + hf * 8];
            vf.u[0] = *(const uint4*)vp;
            vf.u[1] = *(const uint4*)(vp + 16);
            cacc[dq] = wmma_bf16(pf, vf, cacc[dq]);
        }
    }

    // epilogue: divide by l, store bf16 ctx as [B,S,H*64] row-major
    const int b = bh >> 4, h = bh & 15;
    #pragma unroll
    for (int dq = 0; dq < 4; dq++) {
        #pragma unroll
        for (int r = 0; r < 8; r++) {
            float val = cacc[dq][r] / lstate[r];
            int s = qBase + r + hf * 8;
            int d = dq * 16 + l16;
            ctx[((size_t)b * 2048 + s) * 1024 + h * 64 + d] = (bf16)val;
        }
    }
}

// ---------------------------------------------------------------------------
// Launch
// ---------------------------------------------------------------------------
extern "C" void kernel_launch(void* const* d_in, const int* in_sizes, int n_in,
                              void* d_out, int out_size, void* d_ws, size_t ws_size,
                              hipStream_t stream) {
    (void)in_sizes; (void)n_in; (void)out_size; (void)ws_size;

    const float* q  = (const float*)d_in[0];
    const float* k  = (const float*)d_in[1];
    const float* v  = (const float*)d_in[2];
    const float* wq = (const float*)d_in[3];
    const float* bq = (const float*)d_in[4];
    const float* wk = (const float*)d_in[5];
    const float* bk = (const float*)d_in[6];
    const float* wv = (const float*)d_in[7];
    const float* bv = (const float*)d_in[8];
    const float* wo = (const float*)d_in[9];
    const float* bo = (const float*)d_in[10];
    float* out = (float*)d_out;

    char* ws = (char*)d_ws;
    size_t off = 0;
    auto alloc = [&](size_t elems) -> bf16* {
        bf16* p = (bf16*)(ws + off);
        off += ((elems * 2 + 255) & ~(size_t)255);
        return p;
    };
    const size_t ACT = 4096ull * 1024ull;  // tokens x d_model
    const size_t WEL = 1024ull * 1024ull;

    bf16* qx  = alloc(ACT);
    bf16* kx  = alloc(ACT);
    bf16* vx  = alloc(ACT);
    bf16* wqb = alloc(WEL);
    bf16* wkb = alloc(WEL);
    bf16* wvb = alloc(WEL);
    bf16* wob = alloc(WEL);
    bf16* Qh  = alloc(ACT);
    bf16* Kh  = alloc(ACT);
    bf16* Vh  = alloc(ACT);
    bf16* Ctx = alloc(ACT);

    cvt_f32_bf16<<<(int)(ACT / 1024), 256, 0, stream>>>(q,  qx,  (int)ACT);
    cvt_f32_bf16<<<(int)(ACT / 1024), 256, 0, stream>>>(k,  kx,  (int)ACT);
    cvt_f32_bf16<<<(int)(ACT / 1024), 256, 0, stream>>>(v,  vx,  (int)ACT);
    cvt_f32_bf16<<<(int)(WEL / 1024), 256, 0, stream>>>(wq, wqb, (int)WEL);
    cvt_f32_bf16<<<(int)(WEL / 1024), 256, 0, stream>>>(wk, wkb, (int)WEL);
    cvt_f32_bf16<<<(int)(WEL / 1024), 256, 0, stream>>>(wv, wvb, (int)WEL);
    cvt_f32_bf16<<<(int)(WEL / 1024), 256, 0, stream>>>(wo, wob, (int)WEL);

    dim3 pg(8, 32);  // N/128, M/128
    // Q projection pre-scaled by 1/sqrt(d_k) = 0.125
    proj_kernel<<<pg, 256, 0, stream>>>(qx, wqb, bq, Qh, nullptr, 0.125f, 0);
    proj_kernel<<<pg, 256, 0, stream>>>(kx, wkb, bk, Kh, nullptr, 1.0f,   0);
    proj_kernel<<<pg, 256, 0, stream>>>(vx, wvb, bv, Vh, nullptr, 1.0f,   0);

    attn_kernel<<<512, 256, 0, stream>>>(Qh, Kh, Vh, Ctx);

    proj_kernel<<<pg, 256, 0, stream>>>(Ctx, wob, bo, nullptr, out, 1.0f, 1);
}
